// Loss_40750649704757
// MI455X (gfx1250) — matrix operand order repaired and checked by reading, not compiled, not run on hardware
//
#include <hip/hip_runtime.h>
#include <hip/hip_bf16.h>

// ---------------------------------------------------------------------------
// Problem constants (match reference)
// ---------------------------------------------------------------------------
#define NN   4096      // contrastive batch
#define DD   128       // embed dim
#define MM   2048      // pseudo samples
#define KK   1024      // exclusion set
#define INV_TEMP 2.0f                 // 1/TEMP_F, TEMP_F = 0.5
#define SMALL_NUM (-103.61632918f)    // ln(1e-45)
#define INV2S2 0.125f                 // 1/(2*sigma^2), sigma=2
#define LSE_MAX 4.0f                  // analytic bound on neg_sim entries

typedef __attribute__((ext_vector_type(16))) __bf16 v16bf;
typedef __attribute__((ext_vector_type(8)))  __bf16 v8bf;
typedef __attribute__((ext_vector_type(8)))  float  v8f;

// ---------------------------------------------------------------------------
// Helpers
// ---------------------------------------------------------------------------
__device__ __forceinline__ unsigned short f2bf(float f) {
    unsigned u = __float_as_uint(f);
    unsigned r = (u >> 16) & 1u;
    u += 0x7FFFu + r;                  // round-to-nearest-even
    return (unsigned short)(u >> 16);
}

// A fragment: 16x32 bf16 tile (rows rowbase..rowbase+15, K = kbase..kbase+31)
__device__ __forceinline__ v16bf load_a_frag(const __bf16* A, int rowbase,
                                             int lane, int kbase) {
    int row = rowbase + (lane & 15);
    int kb  = (lane >= 16) ? 8 : 0;
    const __bf16* p = A + (size_t)row * DD + kbase;
    v8bf lo = *reinterpret_cast<const v8bf*>(p + kb);       // K = kb .. kb+7
    v8bf hi = *reinterpret_cast<const v8bf*>(p + 16 + kb);  // K = 16+kb .. 16+kb+7
    v16bf r;
#pragma unroll
    for (int i = 0; i < 8; ++i) { r[i] = lo[i]; r[8 + i] = hi[i]; }
    return r;
}

// B fragment for X*Y^T: column n = row n of Y; K = kb..kb+7, 16+kb..16+kb+7.
__device__ __forceinline__ v16bf load_b_frag(const __bf16* Y, int colbase,
                                             int lane, int kbase) {
    int n   = colbase + (lane & 15);
    int kb2 = (lane >= 16) ? 8 : 0;
    const __bf16* p = Y + (size_t)n * DD + kbase;
    v8bf lo = *reinterpret_cast<const v8bf*>(p + kb2);
    v8bf hi = *reinterpret_cast<const v8bf*>(p + 16 + kb2);
    v16bf r;
#pragma unroll
    for (int i = 0; i < 8; ++i) { r[i] = lo[i]; r[8 + i] = hi[i]; }
    return r;
}

// ---------------------------------------------------------------------------
// Kernel: zero the scalar accumulators
// ---------------------------------------------------------------------------
__global__ void init_acc_kernel(float* acc) {
    if (threadIdx.x < 8) acc[threadIdx.x] = 0.0f;
}

// ---------------------------------------------------------------------------
// Kernel: row L2-normalize (eps 1e-8) + convert to bf16.  1 block = 1 row.
// ---------------------------------------------------------------------------
__global__ void normalize_bf16_kernel(const float* __restrict__ x,
                                      unsigned short* __restrict__ out) {
    int row = blockIdx.x, t = threadIdx.x;
    float v  = x[(size_t)row * DD + t];
    float sq = v * v;
#pragma unroll
    for (int msk = 16; msk >= 1; msk >>= 1) sq += __shfl_xor(sq, msk, 32);
    __shared__ float part[4];
    if ((t & 31) == 0) part[t >> 5] = sq;
    __syncthreads();
    float tot = part[0] + part[1] + part[2] + part[3];
    float inv = 1.0f / fmaxf(sqrtf(tot), 1e-8f);
    out[(size_t)row * DD + t] = f2bf(v * inv);
}

// ---------------------------------------------------------------------------
// Kernel: row squared-norm + bf16 copy.  1 block = 1 row.
// ---------------------------------------------------------------------------
__global__ void rows_sq_bf16_kernel(const float* __restrict__ x,
                                    unsigned short* __restrict__ out,
                                    float* __restrict__ sq) {
    int row = blockIdx.x, t = threadIdx.x;
    float v = x[(size_t)row * DD + t];
    float p = v * v;
#pragma unroll
    for (int msk = 16; msk >= 1; msk >>= 1) p += __shfl_xor(p, msk, 32);
    __shared__ float part[4];
    if ((t & 31) == 0) part[t >> 5] = p;
    __syncthreads();
    if (t == 0) sq[row] = part[0] + part[1] + part[2] + part[3];
    out[(size_t)row * DD + t] = f2bf(v);
}

// ---------------------------------------------------------------------------
// Kernel: fused feature loss.
//   block = one 16-row strip; 8 waves each stream disjoint 16-col tiles.
//   Per tile: preload all B fragments + masks, then 8 back-to-back
//   v_wmma_f32_16x16x32_bf16, then branch-free mask/logsumexp epilogue.
//   Row logsumexp = LSE_MAX + log(sum_j exp(x_j - LSE_MAX)); entries are either
//   in [-4,4] or SMALL_NUM/TEMP (~-207, underflows to 0 — negligible).
// ---------------------------------------------------------------------------
__global__ __launch_bounds__(256)
void feature_loss_kernel(const unsigned short* __restrict__ an_us,
                         const unsigned short* __restrict__ bn_us,
                         const float* __restrict__ y_pse,
                         const float* __restrict__ neg_w,
                         const int*   __restrict__ w,
                         float* acc_pos, float* acc_neg) {
    const __bf16* an = reinterpret_cast<const __bf16*>(an_us);
    const __bf16* bn = reinterpret_cast<const __bf16*>(bn_us);
    int lane = threadIdx.x & 31;
    int wave = threadIdx.x >> 5;
    int rowbase = blockIdx.x * 16;
    int rl   = (lane >= 16) ? 8 : 0;
    int colL = lane & 15;

    v16bf a[4];
#pragma unroll
    for (int kc = 0; kc < 4; ++kc) a[kc] = load_a_frag(an, rowbase, lane, kc * 32);

    float s[8];
#pragma unroll
    for (int r = 0; r < 8; ++r) s[r] = 0.0f;
    float pos = 0.0f;

    for (int jt = wave; jt < NN / 16; jt += 8) {
        int colbase = jt * 16;
        // 1) all B fragments for both grams (one big load clause)
        v16bf bb[4], ba[4];
#pragma unroll
        for (int kc = 0; kc < 4; ++kc) {
            bb[kc] = load_b_frag(bn, colbase, lane, kc * 32);
            ba[kc] = load_b_frag(an, colbase, lane, kc * 32);
        }
        // 2) mask loads issued now; latency hidden under the WMMA chain
        int col = colbase + colL;
        size_t base = (size_t)(rowbase + rl) * NN + col;
        float yv[8], nv[8];
        int   wv[8];
#pragma unroll
        for (int r = 0; r < 8; ++r) {
            size_t idx = base + (size_t)r * NN;
            yv[r] = y_pse[idx];
            nv[r] = neg_w[idx];
            wv[r] = w[idx];
        }
        // 3) dense WMMA burst (alternating accumulators -> no RAW chain)
        v8f cc = {}, ci = {};
#pragma unroll
        for (int kc = 0; kc < 4; ++kc) {
            cc = __builtin_amdgcn_wmma_f32_16x16x32_bf16(false, a[kc], false, bb[kc],
                                                         (short)0, cc, false, false);
            ci = __builtin_amdgcn_wmma_f32_16x16x32_bf16(false, a[kc], false, ba[kc],
                                                         (short)0, ci, false, false);
        }
        // 4) branch-free epilogue
#pragma unroll
        for (int r = 0; r < 8; ++r) {
            int row = rowbase + r + rl;
            float cross = cc[r] * INV_TEMP;   // cos/TEMP_F
            float inter = ci[r] * INV_TEMP;
            bool wmask = (wv[r] != 0) || (row == col);
            pos += (wmask && (yv[r] > 0.0f)) ? yv[r] * cross : 0.0f;
            bool negm = (!wmask) && (yv[r] == 0.0f);
            float nwc = fmaxf(nv[r], 0.0f);
            float tc = negm ? nwc * cross : 0.0f;
            tc = (tc == 0.0f) ? SMALL_NUM : tc;
            float ti = negm ? nwc * inter : 0.0f;
            ti = (ti == 0.0f) ? SMALL_NUM : ti;
            s[r] += __expf(tc * INV_TEMP - LSE_MAX)
                  + __expf(ti * INV_TEMP - LSE_MAX);
        }
    }

    // sum partial exp-sums across the 16 lanes that share each row
#pragma unroll
    for (int r = 0; r < 8; ++r) {
#pragma unroll
        for (int msk = 1; msk < 16; msk <<= 1) s[r] += __shfl_xor(s[r], msk, 32);
    }
    __shared__ float ls[8][16];
    if ((lane & 15) == 0) {
#pragma unroll
        for (int r = 0; r < 8; ++r) ls[wave][rl + r] = s[r];
    }
    // positive-loss workgroup partial
#pragma unroll
    for (int msk = 1; msk < 32; msk <<= 1) pos += __shfl_xor(pos, msk, 32);
    if (lane == 0) atomicAdd(acc_pos, pos);
    __syncthreads();
    if (wave == 0) {
        float tot = 0.0f;
        if (lane < 16) {
            float ss = 0.0f;
            for (int wv8 = 0; wv8 < 8; ++wv8) ss += ls[wv8][lane];
            tot = LSE_MAX + __logf(ss);       // row logsumexp
        }
#pragma unroll
        for (int msk = 1; msk < 32; msk <<= 1) tot += __shfl_xor(tot, msk, 32);
        if (lane == 0) atomicAdd(acc_neg, tot);
    }
}

// ---------------------------------------------------------------------------
// Kernel: fused RBF-Gram sum:  accum += sum_ij exp(-max(|ai|^2+|bj|^2-2 ai.bj,0)/8)
//   grid.x = nB/16, grid.y = nA/128; wave -> one 16x16 tile via WMMA.
// ---------------------------------------------------------------------------
__global__ __launch_bounds__(256)
void rbf_sum_kernel(const unsigned short* __restrict__ A_us,
                    const unsigned short* __restrict__ B_us,
                    const float* __restrict__ sqA,
                    const float* __restrict__ sqB,
                    float* accum) {
    const __bf16* A = reinterpret_cast<const __bf16*>(A_us);
    const __bf16* B = reinterpret_cast<const __bf16*>(B_us);
    int lane = threadIdx.x & 31;
    int wave = threadIdx.x >> 5;
    int rowbase = (blockIdx.y * 8 + wave) * 16;
    int colbase = blockIdx.x * 16;

    // preload all fragments -> one load clause, then dense WMMA chain
    v16bf af[4], bf[4];
#pragma unroll
    for (int kc = 0; kc < 4; ++kc) {
        af[kc] = load_a_frag(A, rowbase, lane, kc * 32);
        bf[kc] = load_b_frag(B, colbase, lane, kc * 32);
    }
    int rl  = (lane >= 16) ? 8 : 0;
    int col = colbase + (lane & 15);
    float sb = sqB[col];
    float sa[8];
#pragma unroll
    for (int r = 0; r < 8; ++r) sa[r] = sqA[rowbase + r + rl];

    v8f acc = {};
#pragma unroll
    for (int kc = 0; kc < 4; ++kc) {
        acc = __builtin_amdgcn_wmma_f32_16x16x32_bf16(false, af[kc], false, bf[kc],
                                                      (short)0, acc, false, false);
    }
    float sum = 0.0f;
#pragma unroll
    for (int r = 0; r < 8; ++r) {
        float d = fmaxf(sa[r] + sb - 2.0f * acc[r], 0.0f);
        sum += __expf(-d * INV2S2);
    }
#pragma unroll
    for (int msk = 1; msk < 32; msk <<= 1) sum += __shfl_xor(sum, msk, 32);
    if (lane == 0) atomicAdd(accum, sum);
}

// ---------------------------------------------------------------------------
// Kernel: combine scalars -> final loss
//   acc: [0]=pos_sum [1]=neg_lse_sum [2]=sum_xx [3]=sum_yy [4]=sum_xy [5]=sum_pp
// ---------------------------------------------------------------------------
__global__ void finalize_kernel(const float* acc, float* out) {
    float feat = (-acc[0] + acc[1]) / (float)NN;
    float mmd  = (acc[2] - (float)NN) / ((float)NN * (float)(NN - 1))
               + (acc[3] - (float)MM) / ((float)MM * (float)(MM - 1))
               - 2.0f * acc[4] / ((float)NN * (float)MM);
    float excl = acc[5] - (float)KK;
    out[0] = feat + mmd + excl;
}

// ---------------------------------------------------------------------------
// Launch
// ---------------------------------------------------------------------------
extern "C" void kernel_launch(void* const* d_in, const int* in_sizes, int n_in,
                              void* d_out, int out_size, void* d_ws, size_t ws_size,
                              hipStream_t stream) {
    const float* zi      = (const float*)d_in[0];
    const float* z       = (const float*)d_in[1];
    const float* y_pse   = (const float*)d_in[2];
    const float* neg_w   = (const float*)d_in[3];
    const float* real    = (const float*)d_in[4];
    const float* pseudo  = (const float*)d_in[5];
    const float* pseudos = (const float*)d_in[6];
    const int*   w       = (const int*)d_in[7];
    float* out = (float*)d_out;

    char* ws = (char*)d_ws;
    float* acc = (float*)ws;                                   // 8 accumulators
    unsigned short* anb      = (unsigned short*)(ws + 256);
    unsigned short* bnb      = (unsigned short*)(ws + 256 + 1048576);
    unsigned short* realb    = (unsigned short*)(ws + 256 + 2 * 1048576);
    unsigned short* pseudob  = (unsigned short*)(ws + 256 + 3 * 1048576);
    unsigned short* pseudosb = (unsigned short*)(ws + 256 + 3 * 1048576 + 524288);
    char* tail = ws + 256 + 3 * 1048576 + 524288 + 262144;
    float* sqR  = (float*)tail;
    float* sqP  = (float*)(tail + 16384);
    float* sqPs = (float*)(tail + 16384 + 8192);

    init_acc_kernel<<<1, 32, 0, stream>>>(acc);
    normalize_bf16_kernel<<<NN, DD, 0, stream>>>(zi, anb);
    normalize_bf16_kernel<<<NN, DD, 0, stream>>>(z, bnb);
    rows_sq_bf16_kernel<<<NN, DD, 0, stream>>>(real, realb, sqR);
    rows_sq_bf16_kernel<<<MM, DD, 0, stream>>>(pseudo, pseudob, sqP);
    rows_sq_bf16_kernel<<<KK, DD, 0, stream>>>(pseudos, pseudosb, sqPs);

    feature_loss_kernel<<<NN / 16, 256, 0, stream>>>(anb, bnb, y_pse, neg_w, w,
                                                     &acc[0], &acc[1]);

    rbf_sum_kernel<<<dim3(NN / 16, NN / 128), 256, 0, stream>>>(realb, realb, sqR, sqR, &acc[2]);
    rbf_sum_kernel<<<dim3(MM / 16, MM / 128), 256, 0, stream>>>(pseudob, pseudob, sqP, sqP, &acc[3]);
    rbf_sum_kernel<<<dim3(MM / 16, NN / 128), 256, 0, stream>>>(realb, pseudob, sqR, sqP, &acc[4]);
    rbf_sum_kernel<<<dim3(KK / 16, KK / 128), 256, 0, stream>>>(pseudosb, pseudosb, sqPs, sqPs, &acc[5]);

    finalize_kernel<<<1, 1, 0, stream>>>(acc, out);
}
